// BailingMoELinearAttention_721554506404
// MI455X (gfx1250) — matrix operand order repaired and checked by reading, not compiled, not run on hardware
//
#include <hip/hip_runtime.h>
#include <math.h>

typedef __bf16 bf16_t;
typedef __attribute__((ext_vector_type(16))) __bf16 v16bf;
typedef __attribute__((ext_vector_type(8)))  float  v8f;

#define TOK     8192
#define HIDD    2048
#define NH      16
#define HD      128
#define INNERD  2048
#define NCHUNK  64
#define EPSF    1e-5f
#define SCALEF  0.08838834764831845f   /* 128^-0.5 */
#define ROPEB   600000.0f
#define LDP     136                    /* padded LDS row stride (halves), 16B aligned */

// ---------------------------------------------------------------------------
// WMMA helpers (CDNA5 layouts, ISA 7.12.2)
// ---------------------------------------------------------------------------
__device__ __forceinline__ v8f wmma_bf16(v16bf a, v16bf b, v8f c) {
  return __builtin_amdgcn_wmma_f32_16x16x32_bf16(false, a, false, b, (short)0, c,
                                                 false, false);
}

union FragU { v16bf v; uint4 q[2]; bf16_t h[16]; };

// A fragment: 16x32 bf16, source row-major with stride lda (halves).
// lane<16: row l, halves {k0..k0+7, k0+16..k0+23}; lane>=16: row l-16, halves {+8, +24}.
__device__ __forceinline__ v16bf load_a_frag(const bf16_t* base, int lda, int row0,
                                             int k0, int lane) {
  int l15 = lane & 15, hi = lane >> 4;
  const bf16_t* p = base + (size_t)(row0 + l15) * lda + k0;
  FragU u;
  u.q[0] = *(const uint4*)(p + hi * 8);
  u.q[1] = *(const uint4*)(p + 16 + hi * 8);
  return u.v;
}

// B fragment: 32x16 bf16; source is B^T row-major (row n = column n of B), stride ldb.
// lane<16: K {0..7, 16..23}; lane>=16: K {8..15, 24..31} (matches A fragment).
__device__ __forceinline__ v16bf load_b_frag(const bf16_t* base, int ldb, int col0,
                                             int k0, int lane) {
  int l15 = lane & 15, hi = lane >> 4;
  const bf16_t* p = base + (size_t)(col0 + l15) * ldb + k0;
  FragU u;
  u.q[0] = *(const uint4*)(p + hi * 8);
  u.q[1] = *(const uint4*)(p + 16 + hi * 8);
  return u.v;
}

// CDNA5 async direct-to-LDS copy (ASYNCcnt-tracked); ldsoff = LDS byte offset
// (low 32 bits of a generic pointer to __shared__), gaddr = 64-bit global addr.
__device__ __forceinline__ void async_copy_b128(uint32_t ldsoff, const void* gaddr) {
  asm volatile("global_load_async_to_lds_b128 %0, %1, off"
               :: "v"(ldsoff), "v"(gaddr) : "memory");
}
__device__ __forceinline__ void wait_async0() {
  asm volatile("s_wait_asynccnt 0x0" ::: "memory");
}

// ---------------------------------------------------------------------------
// dtype conversion / weight transpose
// ---------------------------------------------------------------------------
__global__ __launch_bounds__(256) void f32_to_bf16_kernel(const float* __restrict__ src,
                                                          bf16_t* __restrict__ dst,
                                                          size_t n) {
  size_t i = (size_t)blockIdx.x * 256 + threadIdx.x;
  if (i < n) dst[i] = (bf16_t)src[i];
}

// src: K x N f32 row-major  ->  dst: N x K bf16 row-major (B^T layout for WMMA)
__global__ __launch_bounds__(256) void transpose_bf16_kernel(const float* __restrict__ src,
                                                             bf16_t* __restrict__ dst,
                                                             int K, int N) {
  __shared__ float tl[32][33];
  int bn = blockIdx.x * 32, bk = blockIdx.y * 32;
  int tx = threadIdx.x & 31, ty = threadIdx.x >> 5;  // 8 rows per pass
  for (int i = 0; i < 32; i += 8)
    tl[ty + i][tx] = src[(size_t)(bk + ty + i) * N + bn + tx];
  __syncthreads();
  for (int i = 0; i < 32; i += 8)
    dst[(size_t)(bn + ty + i) * K + bk + tx] = (bf16_t)tl[tx][ty + i];
}

// ---------------------------------------------------------------------------
// WMMA GEMM: C[M,N] = act(A[M,K] @ B[K,N]),  A bf16 row-major, BT bf16 (N x K).
// Block tile 128x128, K-step 32, 8 waves each own a 64x32 region (4x2 frags).
// Global->LDS staging uses async direct-to-LDS copies, double-buffered so the
// next K-tile's DMA overlaps the current tile's WMMAs (one barrier per step).
// act: 1 = silu.  Output: Cf (f32) if non-null else Cb (bf16).
// ---------------------------------------------------------------------------
__global__ __launch_bounds__(256) void gemm_bf16_kernel(const bf16_t* __restrict__ A,
                                                        const bf16_t* __restrict__ BT,
                                                        bf16_t* __restrict__ Cb,
                                                        float* __restrict__ Cf,
                                                        int M, int N, int K, int act) {
  __shared__ bf16_t As[2][128 * 40];
  __shared__ bf16_t Bs[2][128 * 40];
  const int tid = threadIdx.x;
  const int lane = tid & 31;
  const int wave = tid >> 5;
  const int wr = wave >> 2;       // 0..1
  const int wc = wave & 3;        // 0..3
  const int bm = blockIdx.x * 128;
  const int bn = blockIdx.y * 128;

  // staging coords: 128 rows x 32 halves per tile, 4 x 16B segs per row,
  // 256 threads -> 2 (row,seg) pairs each
  const int row0 = tid >> 2, seg0 = tid & 3;
  const int row1 = (tid + 256) >> 2, seg1 = tid & 3;

  auto issue = [&](int buf, int kb) {
    async_copy_b128((uint32_t)(uintptr_t)(&As[buf][row0 * 40 + seg0 * 8]),
                    A + (size_t)(bm + row0) * K + kb + seg0 * 8);
    async_copy_b128((uint32_t)(uintptr_t)(&Bs[buf][row0 * 40 + seg0 * 8]),
                    BT + (size_t)(bn + row0) * K + kb + seg0 * 8);
    async_copy_b128((uint32_t)(uintptr_t)(&As[buf][row1 * 40 + seg1 * 8]),
                    A + (size_t)(bm + row1) * K + kb + seg1 * 8);
    async_copy_b128((uint32_t)(uintptr_t)(&Bs[buf][row1 * 40 + seg1 * 8]),
                    BT + (size_t)(bn + row1) * K + kb + seg1 * 8);
  };

  v8f acc[4][2];
  for (int i = 0; i < 4; ++i)
    for (int j = 0; j < 2; ++j)
      for (int r = 0; r < 8; ++r) acc[i][j][r] = 0.f;

  const int nk = K >> 5;
  issue(0, 0);
  wait_async0();
  __syncthreads();

  for (int t = 0; t < nk; ++t) {
    const int buf = t & 1;
    if (t + 1 < nk) issue(buf ^ 1, (t + 1) * 32);   // DMA next tile during compute

    v16bf af[4], bfrag[2];
    for (int i = 0; i < 4; ++i)
      af[i] = load_a_frag(&As[buf][0], 40, wr * 64 + i * 16, 0, lane);
    for (int j = 0; j < 2; ++j)
      bfrag[j] = load_b_frag(&Bs[buf][0], 40, wc * 32 + j * 16, 0, lane);
    for (int i = 0; i < 4; ++i)
      for (int j = 0; j < 2; ++j) acc[i][j] = wmma_bf16(af[i], bfrag[j], acc[i][j]);

    wait_async0();      // own prefetch landed
    __syncthreads();    // everyone's prefetch landed / reads of old buf done
  }

  const int l15 = lane & 15;
  const int hi8 = (lane >> 4) * 8;
  for (int i = 0; i < 4; ++i)
    for (int j = 0; j < 2; ++j)
      for (int r = 0; r < 8; ++r) {
        float x = acc[i][j][r];
        if (act == 1) x = x / (1.f + __expf(-x));   // silu
        size_t row = bm + wr * 64 + i * 16 + hi8 + r;
        size_t col = bn + wc * 32 + j * 16 + l15;
        if (Cf) Cf[row * N + col] = x;
        else    Cb[row * N + col] = (bf16_t)x;
      }
}

// ---------------------------------------------------------------------------
// In-place RMSNorm + RoPE on q,k slices of qkv (bf16, T x 6144), q scaled.
// grid (T, H), block 128
// ---------------------------------------------------------------------------
__global__ __launch_bounds__(128) void normrope_kernel(bf16_t* __restrict__ qkv,
                                                       const int* __restrict__ positions,
                                                       const float* __restrict__ q_ln_w,
                                                       const float* __restrict__ k_ln_w) {
  const int t = blockIdx.x, h = blockIdx.y, d = threadIdx.x;
  __shared__ float sq[2][128];
  __shared__ float buf[2][128];
  const size_t base = (size_t)t * (3 * INNERD) + h * HD;
  float qv = (float)qkv[base + d];
  float kv = (float)qkv[base + INNERD + d];
  sq[0][d] = qv * qv;
  sq[1][d] = kv * kv;
  __syncthreads();
  for (int s = 64; s > 0; s >>= 1) {
    if (d < s) { sq[0][d] += sq[0][d + s]; sq[1][d] += sq[1][d + s]; }
    __syncthreads();
  }
  float rq = rsqrtf(sq[0][0] * (1.f / HD) + EPSF);
  float rk = rsqrtf(sq[1][0] * (1.f / HD) + EPSF);
  buf[0][d] = qv * rq * q_ln_w[d];
  buf[1][d] = kv * rk * k_ln_w[d];
  __syncthreads();
  const int j = d & 63;
  const float pos = (float)positions[t];
  const float invf = __powf(ROPEB, -((float)j) * (1.f / 64.f));
  float sn, cs;
  __sincosf(pos * invf, &sn, &cs);
  float oq, ok;
  if (d < 64) {
    oq = buf[0][d] * cs - buf[0][d + 64] * sn;
    ok = buf[1][d] * cs - buf[1][d + 64] * sn;
  } else {
    oq = buf[0][d] * cs + buf[0][d - 64] * sn;
    ok = buf[1][d] * cs + buf[1][d - 64] * sn;
  }
  qkv[base + d]          = (bf16_t)(oq * SCALEF);
  qkv[base + INNERD + d] = (bf16_t)ok;
}

// ---------------------------------------------------------------------------
// Chunked linear attention. One block per head (16 blocks, 256 thr = 8 waves).
// S^T (128x128 f32) persists in WMMA accumulators across the 64-chunk scan.
// Per chunk: scores = tril(q k^T); o = scores@v + q@S; S^T += v^T@k.
// Dynamic LDS: qs, ks, kTs, vTs, scores, STl  (6 x 128 x LDP bf16).
// ---------------------------------------------------------------------------
__global__ __launch_bounds__(256) void linattn_kernel(const bf16_t* __restrict__ qkv,
                                                      float* __restrict__ o) {
  extern __shared__ char smem[];
  bf16_t* qs  = (bf16_t*)smem;
  bf16_t* ks  = qs  + 128 * LDP;
  bf16_t* kTs = ks  + 128 * LDP;
  bf16_t* vTs = kTs + 128 * LDP;
  bf16_t* sc  = vTs + 128 * LDP;
  bf16_t* STl = sc  + 128 * LDP;

  const int h = blockIdx.x;
  const int tid = threadIdx.x, lane = tid & 31, w = tid >> 5;  // wave 0..7
  const int l15 = lane & 15, hi8 = (lane >> 4) * 8;

  v8f ST[8];  // S^T tiles (rows 16w..16w+15 of S^T = e-dim) x 8 column tiles (d-dim)
  for (int n = 0; n < 8; ++n)
    for (int r = 0; r < 8; ++r) ST[n][r] = 0.f;

  const int srow = tid >> 1;            // staging row 0..127
  const int sseg = (tid & 1) * 8;       // 8 segs of 8 halves each

  for (int c = 0; c < NCHUNK; ++c) {
    // 1. dump S^T -> LDS (bf16) for B-fragment consumption (inter-chunk term)
    for (int n = 0; n < 8; ++n)
      for (int r = 0; r < 8; ++r)
        STl[(size_t)(w * 16 + hi8 + r) * LDP + n * 16 + l15] = (bf16_t)ST[n][r];

    // 2. stage q (row-major), k (row-major), k^T, v^T for this chunk
    {
      const size_t gbase = (size_t)(c * 128 + srow) * (3 * INNERD) + h * HD;
      for (int s = 0; s < 8; ++s) {
        int seg = sseg + s;                           // 0..15
        uint4 qv4 = *(const uint4*)(qkv + gbase + seg * 8);
        uint4 kv4 = *(const uint4*)(qkv + gbase + INNERD + seg * 8);
        uint4 vv4 = *(const uint4*)(qkv + gbase + 2 * INNERD + seg * 8);
        *(uint4*)(qs + srow * LDP + seg * 8) = qv4;
        *(uint4*)(ks + srow * LDP + seg * 8) = kv4;
        FragU ku, vu; ku.q[0] = kv4; vu.q[0] = vv4;
        for (int e = 0; e < 8; ++e) {
          int col = seg * 8 + e;
          kTs[(size_t)col * LDP + srow] = ku.h[e];
          vTs[(size_t)col * LDP + srow] = vu.h[e];
        }
      }
    }
    __syncthreads();

    // 3. scores tiles (w, n): intra-chunk causal q k^T, masked, stored bf16
    for (int n = 0; n < 8; ++n) {
      if (n > w) {
        for (int r = 0; r < 8; ++r)
          sc[(size_t)(w * 16 + hi8 + r) * LDP + n * 16 + l15] = (bf16_t)0.f;
      } else {
        v8f a; for (int r = 0; r < 8; ++r) a[r] = 0.f;
        for (int kb = 0; kb < 4; ++kb) {
          v16bf af = load_a_frag(qs, LDP, w * 16, kb * 32, lane);
          v16bf bfrag = load_b_frag(ks, LDP, n * 16, kb * 32, lane);
          a = wmma_bf16(af, bfrag, a);
        }
        for (int r = 0; r < 8; ++r) {
          int M = w * 16 + hi8 + r;     // c index
          int Nc = n * 16 + l15;        // s index
          float val = (Nc <= M) ? a[r] : 0.f;   // tril mask
          sc[(size_t)M * LDP + Nc] = (bf16_t)val;
        }
      }
    }

    // 4. o tiles (w, n): scores@v + q@S ; write f32 to global
    for (int n = 0; n < 8; ++n) {
      v8f a; for (int r = 0; r < 8; ++r) a[r] = 0.f;
      for (int kb = 0; kb < 4; ++kb) {             // intra: K over s
        v16bf af = load_a_frag(sc, LDP, w * 16, kb * 32, lane);
        v16bf bfrag = load_b_frag(vTs, LDP, n * 16, kb * 32, lane);
        a = wmma_bf16(af, bfrag, a);
      }
      for (int kb = 0; kb < 4; ++kb) {             // inter: K over d, B = S (B^T = S^T)
        v16bf af = load_a_frag(qs, LDP, w * 16, kb * 32, lane);
        v16bf bfrag = load_b_frag(STl, LDP, n * 16, kb * 32, lane);
        a = wmma_bf16(af, bfrag, a);
      }
      for (int r = 0; r < 8; ++r) {
        size_t row = (size_t)c * 128 + w * 16 + hi8 + r;
        size_t col = (size_t)h * HD + n * 16 + l15;
        o[row * INNERD + col] = a[r];
      }
    }

    // 5. state update: S^T += v^T @ k   (A from vTs rows, B^T = k^T rows)
    for (int n = 0; n < 8; ++n)
      for (int kb = 0; kb < 4; ++kb) {
        v16bf af = load_a_frag(vTs, LDP, w * 16, kb * 32, lane);
        v16bf bfrag = load_b_frag(kTs, LDP, n * 16, kb * 32, lane);
        ST[n] = wmma_bf16(af, bfrag, ST[n]);
      }
    __syncthreads();   // protect LDS before next chunk overwrites
  }
}

// ---------------------------------------------------------------------------
// gated = sigmoid(gate) * rmsnorm(o, g_norm_w)  -> bf16.  grid T, block 256
// ---------------------------------------------------------------------------
__global__ __launch_bounds__(256) void gate_kernel(const float* __restrict__ o,
                                                   const bf16_t* __restrict__ gate,
                                                   const float* __restrict__ g_norm_w,
                                                   bf16_t* __restrict__ gated) {
  const int t = blockIdx.x, tid = threadIdx.x;
  __shared__ float red[256];
  float ss = 0.f;
  for (int i = tid; i < INNERD; i += 256) {
    float v = o[(size_t)t * INNERD + i];
    ss += v * v;
  }
  red[tid] = ss;
  __syncthreads();
  for (int s = 128; s > 0; s >>= 1) {
    if (tid < s) red[tid] += red[tid + s];
    __syncthreads();
  }
  float inv = rsqrtf(red[0] * (1.f / INNERD) + EPSF);
  for (int i = tid; i < INNERD; i += 256) {
    float g = (float)gate[(size_t)t * INNERD + i];
    float sig = 1.f / (1.f + __expf(-g));
    float v = o[(size_t)t * INNERD + i] * inv * g_norm_w[i];
    gated[(size_t)t * INNERD + i] = (bf16_t)(sig * v);
  }
}

// ---------------------------------------------------------------------------
extern "C" void kernel_launch(void* const* d_in, const int* in_sizes, int n_in,
                              void* d_out, int out_size, void* d_ws, size_t ws_size,
                              hipStream_t stream) {
  (void)in_sizes; (void)n_in; (void)out_size; (void)ws_size;
  const float* x      = (const float*)d_in[0];
  const int*   pos    = (const int*)d_in[1];
  const float* Wqkv   = (const float*)d_in[2];
  const float* q_ln_w = (const float*)d_in[3];
  const float* k_ln_w = (const float*)d_in[4];
  const float* Wg     = (const float*)d_in[5];
  const float* g_nw   = (const float*)d_in[6];
  const float* Wo     = (const float*)d_in[7];
  float* out = (float*)d_out;

  char* p = (char*)d_ws;
  auto alloc = [&](size_t bytes) {
    char* r = p;
    p += (bytes + 255) & ~(size_t)255;
    return r;
  };
  bf16_t* xb     = (bf16_t*)alloc((size_t)TOK * HIDD * 2);
  bf16_t* WqkvT  = (bf16_t*)alloc((size_t)3 * INNERD * HIDD * 2);
  bf16_t* WgT    = (bf16_t*)alloc((size_t)INNERD * HIDD * 2);
  bf16_t* WoT    = (bf16_t*)alloc((size_t)HIDD * INNERD * 2);
  bf16_t* qkv    = (bf16_t*)alloc((size_t)TOK * 3 * INNERD * 2);
  float*  obuf   = (float*)alloc((size_t)TOK * INNERD * 4);
  bf16_t* gateb  = (bf16_t*)alloc((size_t)TOK * INNERD * 2);
  bf16_t* gated  = (bf16_t*)alloc((size_t)TOK * INNERD * 2);

  // 1. convert inputs to bf16 (weights transposed to N x K for B-fragments)
  f32_to_bf16_kernel<<<(TOK * HIDD) / 256, 256, 0, stream>>>(x, xb, (size_t)TOK * HIDD);
  transpose_bf16_kernel<<<dim3((3 * INNERD) / 32, HIDD / 32), 256, 0, stream>>>(
      Wqkv, WqkvT, HIDD, 3 * INNERD);
  transpose_bf16_kernel<<<dim3(INNERD / 32, HIDD / 32), 256, 0, stream>>>(
      Wg, WgT, HIDD, INNERD);
  transpose_bf16_kernel<<<dim3(HIDD / 32, INNERD / 32), 256, 0, stream>>>(
      Wo, WoT, INNERD, HIDD);

  // 2. qkv = silu(x @ Wqkv)   [8192 x 6144]
  gemm_bf16_kernel<<<dim3(TOK / 128, (3 * INNERD) / 128), 256, 0, stream>>>(
      xb, WqkvT, qkv, nullptr, TOK, 3 * INNERD, HIDD, 1);

  // 3. per-head RMSNorm + RoPE on q,k (in place), q scaled by D^-1/2
  normrope_kernel<<<dim3(TOK, NH), 128, 0, stream>>>(qkv, pos, q_ln_w, k_ln_w);

  // 4. chunked linear attention (state in WMMA accumulators)
  linattn_kernel<<<NH, 256, 6 * 128 * LDP * sizeof(bf16_t), stream>>>(qkv, obuf);

  // 5. gate = x @ Wg
  gemm_bf16_kernel<<<dim3(TOK / 128, INNERD / 128), 256, 0, stream>>>(
      xb, WgT, gateb, nullptr, TOK, INNERD, HIDD, 0);

  // 6. gated = sigmoid(gate) * rmsnorm(o)
  gate_kernel<<<TOK, 256, 0, stream>>>(obuf, gateb, g_nw, gated);

  // 7. out = gated @ Wo  (f32 output)
  gemm_bf16_kernel<<<dim3(TOK / 128, HIDD / 128), 256, 0, stream>>>(
      gated, WoT, nullptr, out, TOK, HIDD, INNERD, 0);
}
